// GUARDIAN_69312182223528
// MI455X (gfx1250) — matrix operand order repaired and checked by reading, not compiled, not run on hardware
//
#include <hip/hip_runtime.h>
#include <cstdint>
#include <cstddef>

#define NNODES 8000
#define NEDGES 80000
#define EDIM   8
#define HDIM   256
#define HHALF  128
#define G4     512   /* 4*HH gates */

typedef __attribute__((ext_vector_type(16))) __bf16 v16bf;
typedef __attribute__((ext_vector_type(8)))  float  v8f;

union FragBF { uint4 q[2]; v16bf v; };
union FragF  { float4 f[2]; v8f   v; };

static __device__ inline unsigned short f2bf(float f){
  unsigned int u = __builtin_bit_cast(unsigned int, f);
  u += 0x7FFFu + ((u >> 16) & 1u);
  return (unsigned short)(u >> 16);
}
static __device__ inline float bf2f(unsigned short u){
  unsigned int x = ((unsigned int)u) << 16;
  return __builtin_bit_cast(float, x);
}
static __device__ inline float sigmoidf_(float x){ return 1.0f / (1.0f + expf(-x)); }

// A fragment (16Mx32K bf16): per lane = one contiguous 32B run.
static __device__ inline v16bf load_a_frag(const unsigned short* rowbase, int stride, int k0){
  int lane = threadIdx.x & 31;
  int m = lane & 15;
  int khalf = ((lane >> 4) & 1) * 16;
  const unsigned short* r = rowbase + m * stride + k0 + khalf;
  FragBF f;
  f.q[0] = *(const uint4*)(r);
  f.q[1] = *(const uint4*)(r + 8);
  return f.v;
}
// B fragment (32Kx16N bf16) from LDS tile stored [n][32k]: one contiguous 32B run.
static __device__ inline v16bf load_b_frag(const unsigned short* tile, int n0){
  int lane = threadIdx.x & 31;
  int n = lane & 15;
  int kh = ((lane >> 4) & 1) * 16;
  const unsigned short* r = tile + (n0 + n) * 32 + kh;
  FragBF f;
  f.q[0] = *(const uint4*)(r);
  f.q[1] = *(const uint4*)(r + 8);
  return f.v;
}

// ---- CDNA5 async global->LDS copy: 32B per thread (2 x b128), ASYNCcnt tracked ----
static __device__ inline void async_copy32(const void* sbase, int byte_off, void* lds_dst){
  unsigned int lds = (unsigned int)(uintptr_t)lds_dst;  // flat LDS addr low 32 bits = LDS offset
  asm volatile("global_load_async_to_lds_b128 %0, %1, %2\n\t"
               "global_load_async_to_lds_b128 %3, %4, %2"
               :
               : "v"(lds), "v"(byte_off), "s"(sbase),
                 "v"(lds + 16u), "v"(byte_off + 16)
               : "memory");
}
template<int N>
static __device__ inline void wait_async_le(){
  asm volatile("s_wait_asynccnt %0" :: "i"(N) : "memory");
}
static __device__ inline void sched_fence(){
  asm volatile("" ::: "memory");
}

// =================== weight f32 -> bf16 prep ===================
__global__ void cvt_kernel(const float* src, unsigned short* dst, int n){
  int i = blockIdx.x * 256 + threadIdx.x;
  if (i < n) dst[i] = f2bf(src[i]);
}

// =================== bucketing ===================
__global__ void init_kernel(int* cin, int* cout, unsigned int* mm){
  int i = blockIdx.x * 256 + threadIdx.x;
  if (i < NNODES){ cin[i] = 0; cout[i] = 0; }
  if (i == 0){ mm[0] = 0xFFFFFFFFu; mm[1] = 0u; }
}

__global__ void count_kernel(const int* ei, int* cin, int* cout){
  int i = blockIdx.x * 256 + threadIdx.x;
  if (i >= NEDGES) return;
  atomicAdd(&cout[ei[i]], 1);
  atomicAdd(&cin [ei[NEDGES + i]], 1);
}

__global__ void minmax_kernel(const float* ts, unsigned int* mm){
  __shared__ unsigned int smin, smax;
  if (threadIdx.x == 0){ smin = 0xFFFFFFFFu; smax = 0u; }
  __syncthreads();
  int i = blockIdx.x * 256 + threadIdx.x;
  if (i < NEDGES){
    unsigned int b = __builtin_bit_cast(unsigned int, ts[i]); // ts >= 0
    atomicMin(&smin, b); atomicMax(&smax, b);
  }
  __syncthreads();
  if (threadIdx.x == 0){ atomicMin(&mm[0], smin); atomicMax(&mm[1], smax); }
}

__global__ __launch_bounds__(1024) void scan_kernel(const int* counts, int* starts, int* cursor){
  __shared__ int buf[1024];
  __shared__ int carry;
  if (threadIdx.x == 0) carry = 0;
  __syncthreads();
  for (int base = 0; base < NNODES; base += 1024){
    int i = base + threadIdx.x;
    int v = (i < NNODES) ? counts[i] : 0;
    buf[threadIdx.x] = v;
    __syncthreads();
    for (int off = 1; off < 1024; off <<= 1){
      int t = (threadIdx.x >= off) ? buf[threadIdx.x - off] : 0;
      __syncthreads();
      buf[threadIdx.x] += t;
      __syncthreads();
    }
    int excl = buf[threadIdx.x] - v + carry;
    if (i < NNODES){ starts[i] = excl; cursor[i] = excl; }
    int tot = buf[1023];
    __syncthreads();
    if (threadIdx.x == 0) carry += tot;
    __syncthreads();
  }
}

__global__ void fill_kernel(const int* ei, int* cur_in, int* cur_out, int* csr_in, int* csr_out){
  int i = blockIdx.x * 256 + threadIdx.x;
  if (i >= NEDGES) return;
  csr_out[atomicAdd(&cur_out[ei[i]], 1)] = i;
  csr_in [atomicAdd(&cur_in [ei[NEDGES + i]], 1)] = i;
}

__global__ void sort_kernel(const int* starts, const int* lens, int* csr){
  int n = blockIdx.x * 256 + threadIdx.x;
  if (n >= NNODES) return;
  int s = starts[n], L = lens[n];
  for (int i = 1; i < L; ++i){
    int key = csr[s + i]; int j = i - 1;
    while (j >= 0 && csr[s + j] > key){ csr[s + j + 1] = csr[s + j]; --j; }
    csr[s + j + 1] = key;
  }
}

// =================== edge feature projection + positional emb ===================
__global__ __launch_bounds__(256) void ef_kernel(
    const float* attr, const float* ts, const float* Wp, const float* bp,
    const float* pe, const float* tscale, const unsigned int* mm,
    float* ef_f32, unsigned short* ef_bf)
{
  int e = blockIdx.x * 8 + (threadIdx.x >> 5);
  if (e >= NEDGES) return;
  int lane = threadIdx.x & 31;
  float tmin = __builtin_bit_cast(float, mm[0]);
  float tmax = __builtin_bit_cast(float, mm[1]);
  float a[EDIM];
#pragma unroll
  for (int k = 0; k < EDIM; ++k) a[k] = attr[e * EDIM + k];
  int p = 0;
  if (tmax > tmin){
    p = (int)((ts[e] - tmin) / (tmax - tmin) * 4999.0f);
    p = min(max(p, 0), 4999);
  }
  float sc = tscale[0];
  int c0 = lane * 8;
  float out[8];
  unsigned short ob[8];
#pragma unroll
  for (int j = 0; j < 8; ++j){
    int c = c0 + j;
    float acc = bp[c];
#pragma unroll
    for (int k = 0; k < EDIM; ++k) acc += a[k] * Wp[c * EDIM + k];
    acc += sc * pe[(size_t)p * HDIM + c];
    out[j] = acc; ob[j] = f2bf(acc);
  }
  float4* o4 = (float4*)(ef_f32 + (size_t)e * HDIM + c0);
  o4[0] = make_float4(out[0], out[1], out[2], out[3]);
  o4[1] = make_float4(out[4], out[5], out[6], out[7]);
  *(uint4*)(ef_bf + (size_t)e * HDIM + c0) = *(uint4*)ob;
}

// =================== xW precompute: [E,256]bf16 x Wih^T -> [E,512]bf16 ===================
// Double-buffered async pipeline, manually unrolled x2 so each body has a
// compile-time buffer index; N-tiles computed in groups of 4 to cap live B regs.
__global__ __launch_bounds__(256) void gemm_xw_kernel(
    const unsigned short* A, const unsigned short* WihBF,
    const float* bih, const float* bhh, unsigned short* Out)
{
  __shared__ unsigned short Alds[2][128][32];
  __shared__ unsigned short Blds[2][128][32];   // [n][k] = W row layout
  int tid = threadIdx.x;
  int m0 = blockIdx.x * 128;
  int n0 = blockIdx.y * 128;
  int wave = tid >> 5, lane = tid & 31;
  int wm = wave * 16;
  int row = tid >> 1, kb = (tid & 1) * 16;
  v8f acc[8];
#pragma unroll
  for (int t = 0; t < 8; ++t)
#pragma unroll
    for (int i = 0; i < 8; ++i) acc[t][i] = 0.0f;

  auto issue = [&](int k0, int buf){
    async_copy32(A,     ((m0 + row) * HDIM + k0 + kb) * 2, &Alds[buf][row][kb]);
    async_copy32(WihBF, ((n0 + row) * HDIM + k0 + kb) * 2, &Blds[buf][row][kb]);
  };
  auto body = [&](int kt, int buf, int KSTEPS){
    if (kt + 1 < KSTEPS){
      issue((kt + 1) * 32, buf ^ 1);
      wait_async_le<4>();       // tile kt drained; kt+1 stays in flight
    } else {
      wait_async_le<0>();
    }
    __syncthreads();
    v16bf af = load_a_frag(&Alds[buf][wm][0], 32, 0);
#pragma unroll
    for (int nt = 0; nt < 8; ++nt){
      if (nt == 4) sched_fence();  // cap live B fragments at 4
      v16bf bf = load_b_frag(&Blds[buf][0][0], nt * 16);
      acc[nt] = __builtin_amdgcn_wmma_f32_16x16x32_bf16(false, af, false, bf,
                                                        (short)0, acc[nt], false, false);
    }
    __syncthreads();            // compute done before this buffer is refilled
  };

  const int KSTEPS = HDIM / 32;
  issue(0, 0);
#pragma unroll
  for (int kt2 = 0; kt2 < KSTEPS; kt2 += 2){
    body(kt2,     0, KSTEPS);
    body(kt2 + 1, 1, KSTEPS);
  }

  int coln = lane & 15;
  int mb = (lane >> 4) * 8;
#pragma unroll
  for (int nt = 0; nt < 8; ++nt){
    int col = n0 + nt * 16 + coln;
    float bias = bih[col] + bhh[col];
#pragma unroll
    for (int r = 0; r < 8; ++r){
      int orow = m0 + wm + mb + r;
      Out[(size_t)orow * G4 + col] = f2bf(acc[nt][r] + bias);
    }
  }
}

// =================== recurrent BiLSTM: 16 nodes/WG, 32 waves, WMMA M=16 ===================
__global__ __launch_bounds__(1024) void lstm_kernel(
    const int* csr, const int* starts, const int* lens,
    const unsigned short* xw_f, const unsigned short* xw_b,
    const unsigned short* WhhBF_f, const unsigned short* WhhBF_b,
    const float* ef_f32, unsigned short* nodefeat, int colbase)
{
  __shared__ unsigned short h_lds[2][16][HHALF];   // bf16 hidden per dir
  __shared__ float gbuf[32][32][8];                // fragment-native gates, one dir at a time
  __shared__ int s_start[16], s_len[16], s_T;

  int tid = threadIdx.x;
  int nodeBase = blockIdx.x * 16;

  if (tid < 16){ s_start[tid] = starts[nodeBase + tid]; s_len[tid] = lens[nodeBase + tid]; }
  for (int i = tid; i < 2 * 16 * HHALF; i += 1024) ((unsigned short*)h_lds)[i] = 0;
  __syncthreads();
  if (tid == 0){
    int T = 0;
    for (int i = 0; i < 16; ++i) T = max(T, s_len[i]);
    s_T = T;
  }
  __syncthreads();
  int T = s_T;

  // 32 waves: 16 per direction, 2 N-tiles (32 gate cols) each; Whh frags register-resident.
  int wave = tid >> 5, lane = tid & 31;
  int dirW = wave >> 4;
  int n0 = (wave & 15) * 32;
  const unsigned short* WhhW = dirW ? WhhBF_b : WhhBF_f;
  v16bf bfrag[2][4];
  {
    int n = lane & 15;
    int kh = ((lane >> 4) & 1) * 16;
#pragma unroll
    for (int nt = 0; nt < 2; ++nt)
#pragma unroll
      for (int ks = 0; ks < 4; ++ks){
        const unsigned short* src = WhhW + (size_t)(n0 + nt * 16 + n) * HHALF + ks * 32 + kh;
        FragBF f;
        f.q[0] = *(const uint4*)(src);
        f.q[1] = *(const uint4*)(src + 8);
        bfrag[nt][ks] = f.v;
      }
  }

  // nonlinearity mapping: (node, 2 units) per thread, cell state in registers
  int nlNode = tid >> 6;          // 0..15
  int nlU0 = (tid & 63) * 2;      // 2 consecutive units
  float cst[2][2];
#pragma unroll
  for (int d = 0; d < 2; ++d)
#pragma unroll
    for (int j = 0; j < 2; ++j) cst[d][j] = 0.0f;

  for (int s = 0; s < T; ++s){
    for (int d = 0; d < 2; ++d){
      __syncthreads();
      if (dirW == d){                 // wave-uniform: EXEC all-ones for WMMA
        v16bf afrag[4];
#pragma unroll
        for (int ks = 0; ks < 4; ++ks)
          afrag[ks] = load_a_frag(&h_lds[d][0][0], HHALF, ks * 32);
#pragma unroll
        for (int nt = 0; nt < 2; ++nt){
          v8f c;
#pragma unroll
          for (int i = 0; i < 8; ++i) c[i] = 0.0f;
#pragma unroll
          for (int ks = 0; ks < 4; ++ks)
            c = __builtin_amdgcn_wmma_f32_16x16x32_bf16(false, afrag[ks], false, bfrag[nt][ks],
                                                        (short)0, c, false, false);
          int slot = (wave & 15) * 2 + nt;       // 16-col gate slot, 0..31
          FragF ff; ff.v = c;
          float4* dst = (float4*)&gbuf[slot][lane][0];
          dst[0] = ff.f[0];
          dst[1] = ff.f[1];
        }
      }
      __syncthreads();
      { // gate nonlinearity; validity mask == jnp.where(v, h2, h)
        int t = (d == 0) ? s : (T - 1 - s);
        int len = s_len[nlNode];
        if (t < len){
          const unsigned short* xwB = (d == 0) ? xw_f : xw_b;
          int eid = csr[s_start[nlNode] + t];
          const unsigned short* xw = xwB + (size_t)eid * G4;
          // prefetch the next step's gathered gate-input row
          int tn = t + ((d == 0) ? 1 : -1);
          if (tn >= 0 && tn < len)
            __builtin_prefetch(xwB + (size_t)csr[s_start[nlNode] + tn] * G4 + nlU0, 0, 1);
          int r = nlNode & 7;
          int lhalf = (nlNode >= 8) ? 16 : 0;
          unsigned int hpack = 0;
#pragma unroll
          for (int j = 0; j < 2; ++j){
            int u = nlU0 + j;
            int lcol = (u & 15) | lhalf;
            int shi = u >> 4;
            float gi = gbuf[shi     ][lcol][r] + bf2f(xw[u]);
            float gf = gbuf[8  + shi][lcol][r] + bf2f(xw[HHALF + u]);
            float gg = gbuf[16 + shi][lcol][r] + bf2f(xw[2 * HHALF + u]);
            float go = gbuf[24 + shi][lcol][r] + bf2f(xw[3 * HHALF + u]);
            float c2 = sigmoidf_(gf) * cst[d][j] + sigmoidf_(gi) * tanhf(gg);
            float h2 = sigmoidf_(go) * tanhf(c2);
            cst[d][j] = c2;
            hpack |= ((unsigned int)f2bf(h2)) << (16 * j);
          }
          *(unsigned int*)&h_lds[d][nlNode][nlU0] = hpack;
        }
      }
    }
  }
  __syncthreads();
  // finalize: concat(h_fwd, h_bwd) -> bf16 nodefeat; len==1 bypass; len==0 zeros
  for (int i = tid; i < 16 * HDIM; i += 1024){
    int node = i >> 8;
    int u = i & 255;
    unsigned short val;
    int L = s_len[node];
    if (L == 1){
      int eid = csr[s_start[node]];
      val = f2bf(ef_f32[(size_t)eid * HDIM + u]);
    } else {
      val = h_lds[u >> 7][node][u & 127];
    }
    nodefeat[(size_t)(nodeBase + node) * (2 * HDIM) + colbase + u] = val;
  }
}

// =================== fusion: relu([N,512]bf16 x W_fuse^T + b) -> f32 ===================
__global__ __launch_bounds__(256) void fuse_kernel(
    const unsigned short* nodef, const unsigned short* WfBF, const float* bfuse, float* out)
{
  __shared__ unsigned short Alds[2][128][32];
  __shared__ unsigned short Blds[2][128][32];
  int tid = threadIdx.x;
  int m0 = blockIdx.x * 128;
  int n0 = blockIdx.y * 128;
  int wave = tid >> 5, lane = tid & 31;
  int wm = wave * 16;
  int row = tid >> 1, kb = (tid & 1) * 16;
  int rg = min(m0 + row, NNODES - 1);
  v8f acc[8];
#pragma unroll
  for (int t = 0; t < 8; ++t)
#pragma unroll
    for (int i = 0; i < 8; ++i) acc[t][i] = 0.0f;

  auto issue = [&](int k0, int buf){
    async_copy32(nodef, (rg * (2 * HDIM) + k0 + kb) * 2, &Alds[buf][row][kb]);
    async_copy32(WfBF,  ((n0 + row) * (2 * HDIM) + k0 + kb) * 2, &Blds[buf][row][kb]);
  };
  auto body = [&](int kt, int buf, int KSTEPS){
    if (kt + 1 < KSTEPS){
      issue((kt + 1) * 32, buf ^ 1);
      wait_async_le<4>();
    } else {
      wait_async_le<0>();
    }
    __syncthreads();
    v16bf af = load_a_frag(&Alds[buf][wm][0], 32, 0);
#pragma unroll
    for (int nt = 0; nt < 8; ++nt){
      if (nt == 4) sched_fence();
      v16bf bfr = load_b_frag(&Blds[buf][0][0], nt * 16);
      acc[nt] = __builtin_amdgcn_wmma_f32_16x16x32_bf16(false, af, false, bfr,
                                                        (short)0, acc[nt], false, false);
    }
    __syncthreads();
  };

  const int KSTEPS = (2 * HDIM) / 32;
  issue(0, 0);
#pragma unroll
  for (int kt2 = 0; kt2 < KSTEPS; kt2 += 2){
    body(kt2,     0, KSTEPS);
    body(kt2 + 1, 1, KSTEPS);
  }

  int coln = lane & 15;
  int mb = (lane >> 4) * 8;
#pragma unroll
  for (int nt = 0; nt < 8; ++nt){
    int col = n0 + nt * 16 + coln;
    float bias = bfuse[col];
#pragma unroll
    for (int r = 0; r < 8; ++r){
      int orow = m0 + wm + mb + r;
      if (orow < NNODES)
        out[(size_t)orow * (2 * HDIM) + col] = fmaxf(acc[nt][r] + bias, 0.0f);
    }
  }
}

// =================== launch ===================
extern "C" void kernel_launch(void* const* d_in, const int* in_sizes, int n_in,
                              void* d_out, int out_size, void* d_ws, size_t ws_size,
                              hipStream_t stream) {
  (void)in_sizes; (void)n_in; (void)out_size; (void)ws_size;
  const int*   edge_index = (const int*)  d_in[0];
  const float* edge_attr  = (const float*)d_in[1];
  const float* ts         = (const float*)d_in[2];
  const float* W_proj     = (const float*)d_in[3];
  const float* b_proj     = (const float*)d_in[4];
  const float* pos_emb    = (const float*)d_in[5];
  const float* time_scale = (const float*)d_in[6];
  const float* in_Wih     = (const float*)d_in[7];
  const float* in_Whh     = (const float*)d_in[8];
  const float* in_bih     = (const float*)d_in[9];
  const float* in_bhh     = (const float*)d_in[10];
  const float* out_Wih    = (const float*)d_in[11];
  const float* out_Whh    = (const float*)d_in[12];
  const float* out_bih    = (const float*)d_in[13];
  const float* out_bhh    = (const float*)d_in[14];
  const float* W_fuse     = (const float*)d_in[15];
  const float* b_fuse     = (const float*)d_in[16];
  float* out = (float*)d_out;

  char* ws = (char*)d_ws;
  size_t off = 0;
  auto alloc = [&](size_t bytes) -> void* {
    void* p = ws + off;
    off = (off + bytes + 255) & ~(size_t)255;
    return p;
  };
  int* counts_in  = (int*)alloc(NNODES * 4);
  int* counts_out = (int*)alloc(NNODES * 4);
  int* starts_in  = (int*)alloc(NNODES * 4);
  int* starts_out = (int*)alloc(NNODES * 4);
  int* cursor_in  = (int*)alloc(NNODES * 4);
  int* cursor_out = (int*)alloc(NNODES * 4);
  int* csr_in     = (int*)alloc(NEDGES * 4);
  int* csr_out    = (int*)alloc(NEDGES * 4);
  unsigned int* mm = (unsigned int*)alloc(8);
  float*          ef_f32 = (float*)         alloc((size_t)NEDGES * HDIM * 4);
  unsigned short* ef_bf  = (unsigned short*)alloc((size_t)NEDGES * HDIM * 2);
  unsigned short* xw_f   = (unsigned short*)alloc((size_t)NEDGES * G4 * 2);
  unsigned short* xw_b   = (unsigned short*)alloc((size_t)NEDGES * G4 * 2);
  unsigned short* nodef  = (unsigned short*)alloc((size_t)NNODES * 2 * HDIM * 2);
  unsigned short* wbf_in_Wih  = (unsigned short*)alloc((size_t)2 * G4 * HDIM * 2);
  unsigned short* wbf_out_Wih = (unsigned short*)alloc((size_t)2 * G4 * HDIM * 2);
  unsigned short* wbf_in_Whh  = (unsigned short*)alloc((size_t)2 * G4 * HHALF * 2);
  unsigned short* wbf_out_Whh = (unsigned short*)alloc((size_t)2 * G4 * HHALF * 2);
  unsigned short* wbf_fuse    = (unsigned short*)alloc((size_t)2 * HDIM * 2 * HDIM * 2);

  const int EB = (NEDGES + 255) / 256;
  const int NB = (NNODES + 255) / 256;

  // weight conversion (one-time per call)
  cvt_kernel<<<(2 * G4 * HDIM + 255) / 256, 256, 0, stream>>>(in_Wih,  wbf_in_Wih,  2 * G4 * HDIM);
  cvt_kernel<<<(2 * G4 * HDIM + 255) / 256, 256, 0, stream>>>(out_Wih, wbf_out_Wih, 2 * G4 * HDIM);
  cvt_kernel<<<(2 * G4 * HHALF + 255) / 256, 256, 0, stream>>>(in_Whh,  wbf_in_Whh,  2 * G4 * HHALF);
  cvt_kernel<<<(2 * G4 * HHALF + 255) / 256, 256, 0, stream>>>(out_Whh, wbf_out_Whh, 2 * G4 * HHALF);
  cvt_kernel<<<(4 * HDIM * HDIM + 255) / 256, 256, 0, stream>>>(W_fuse, wbf_fuse, 4 * HDIM * HDIM);

  // bucketing + temporal range
  init_kernel<<<NB, 256, 0, stream>>>(counts_in, counts_out, mm);
  count_kernel<<<EB, 256, 0, stream>>>(edge_index, counts_in, counts_out);
  minmax_kernel<<<EB, 256, 0, stream>>>(ts, mm);
  scan_kernel<<<1, 1024, 0, stream>>>(counts_in, starts_in, cursor_in);
  scan_kernel<<<1, 1024, 0, stream>>>(counts_out, starts_out, cursor_out);
  fill_kernel<<<EB, 256, 0, stream>>>(edge_index, cursor_in, cursor_out, csr_in, csr_out);
  sort_kernel<<<NB, 256, 0, stream>>>(starts_in, counts_in, csr_in);
  sort_kernel<<<NB, 256, 0, stream>>>(starts_out, counts_out, csr_out);

  // edge features
  ef_kernel<<<(NEDGES + 7) / 8, 256, 0, stream>>>(edge_attr, ts, W_proj, b_proj,
                                                  pos_emb, time_scale, mm, ef_f32, ef_bf);

  dim3 gXW(NEDGES / 128, G4 / 128);
  // incoming aggregation
  gemm_xw_kernel<<<gXW, 256, 0, stream>>>(ef_bf, wbf_in_Wih,              in_bih,      in_bhh,      xw_f);
  gemm_xw_kernel<<<gXW, 256, 0, stream>>>(ef_bf, wbf_in_Wih + G4 * HDIM,  in_bih + G4, in_bhh + G4, xw_b);
  lstm_kernel<<<NNODES / 16, 1024, 0, stream>>>(csr_in, starts_in, counts_in, xw_f, xw_b,
                                                wbf_in_Whh, wbf_in_Whh + G4 * HHALF, ef_f32, nodef, 0);
  // outgoing aggregation (reuse xw buffers)
  gemm_xw_kernel<<<gXW, 256, 0, stream>>>(ef_bf, wbf_out_Wih,             out_bih,      out_bhh,      xw_f);
  gemm_xw_kernel<<<gXW, 256, 0, stream>>>(ef_bf, wbf_out_Wih + G4 * HDIM, out_bih + G4, out_bhh + G4, xw_b);
  lstm_kernel<<<NNODES / 16, 1024, 0, stream>>>(csr_out, starts_out, counts_out, xw_f, xw_b,
                                                wbf_out_Whh, wbf_out_Whh + G4 * HHALF, ef_f32, nodef, HDIM);

  // fusion + relu
  dim3 gF((NNODES + 127) / 128, (2 * HDIM) / 128);
  fuse_kernel<<<gF, 256, 0, stream>>>(nodef, wbf_fuse, b_fuse, out);
}
